// VectorToVector_51977694216368
// MI455X (gfx1250) — matrix operand order repaired and checked by reading, not compiled, not run on hardware
//
#include <hip/hip_runtime.h>
#include <stddef.h>

// ---------------------------------------------------------------------------
// Rotation-equivariant 5x5 conv as implicit GEMM on CDNA5 WMMA (gfx1250).
//   A : built conv kernel, [Co=128][K], bf16 hi/lo split, K is TAP-MAJOR:
//       k = (kh*5+kw)*128 + ci  -> a K=32 chunk has ONE tap (uniform kh/kw)
//   B : im2col tile of x, staged per K-chunk into LDS (bf16 hi/lo split),
//       branchless OOB handling (clamped base + cndmask-to-zero)
//   D = A x B in fp32 accum via v_wmma_f32_16x16x32_bf16 (bf16x2 split:
//       ah*bh + ah*bl + al*bh)
//   A chunks are DMA'd into LDS by the Tensor Data Mover with LDS padding
//   (80B rows) for bank-conflict-free, 16B-aligned fragment reads;
//   double-buffered pipeline with one barrier per chunk.
// ---------------------------------------------------------------------------

typedef __bf16 bf16_t;
typedef __attribute__((ext_vector_type(16))) __bf16 v16bf;
typedef __attribute__((ext_vector_type(8)))  float  v8f;
typedef unsigned int u32;
typedef __attribute__((ext_vector_type(4))) u32 u32x4;
typedef __attribute__((ext_vector_type(8))) int i32x8;
typedef __attribute__((ext_vector_type(4))) int i32x4;

#define OUT_CH2 128
#define IN_CH2  128
#define KK      5
#define KTAPS   25
#define KDIM    (IN_CH2 * KTAPS)   // 3200
#define HH      128
#define WW      128
#define HW      (HH * WW)
#define BB      16
#define TILE_N  64
#define KC      32
#define NCHUNK  (KDIM / KC)        // 100
#define BSTRIDE 40                 // LDS row stride (bf16 elems) = 80B = 20 dw

#ifndef __has_builtin
#define __has_builtin(x) 0
#endif
#if __has_builtin(__builtin_amdgcn_tensor_load_to_lds) && \
    __has_builtin(__builtin_amdgcn_s_wait_tensorcnt)
#define USE_TDM 1
#else
#define USE_TDM 0
#endif

__device__ __forceinline__ bf16_t f2bf(float f) { return (bf16_t)f; }

__device__ __forceinline__ v16bf load8x2(const bf16_t* p0, const bf16_t* p1) {
  union { v16bf v; uint4 q[2]; } u;
  u.q[0] = *(const uint4*)p0;
  u.q[1] = *(const uint4*)p1;
  return u.v;
}

// Single __shared__ struct -> TDM LDS byte offsets are offsetof()-known.
struct SMem {
#if USE_TDM
  bf16_t Ah[2][OUT_CH2 * BSTRIDE];
  bf16_t Al[2][OUT_CH2 * BSTRIDE];
#endif
  bf16_t Bh[2][TILE_N * BSTRIDE];
  bf16_t Bl[2][TILE_N * BSTRIDE];
};

#if USE_TDM
// TDM 2D tile: 32 elems x 128 rows of bf16, global row stride 3200 elems,
// LDS pad: 4 dwords after every 16 dwords -> 80B LDS rows.
__device__ __forceinline__ void tdm_load_a(const bf16_t* gsrc, u32 lds_off) {
  unsigned long long ga = (unsigned long long)gsrc;
  u32x4 g0;
  g0[0] = 1u;                                   // count=1, user mode
  g0[1] = lds_off;                              // lds_addr (bytes)
  g0[2] = (u32)ga;                              // global_addr[31:0]
  g0[3] = (u32)(ga >> 32) | (2u << 30);         // global_addr[56:32] | type=2
  i32x8 g1;
  g1[0] = (int)((1u << 16) |                    // data_size = 2B
                (1u << 20) |                    // pad_enable
                (3u << 22) |                    // pad_interval = 16 dwords
                (3u << 25));                    // pad_amount   = 4 dwords
  g1[1] = (int)(3200u << 16);                   // tensor_dim0[15:0] << 16
  g1[2] = (int)(128u << 16);                    // tensor_dim1 lo << 16
  g1[3] = (int)(32u << 16);                     // tile_dim0 = 32
  g1[4] = (int)128;                             // tile_dim1 = 128
  g1[5] = (int)3200;                            // tensor_dim0_stride
  g1[6] = 0;
  g1[7] = 0;
  i32x4 z4 = {0, 0, 0, 0};
#if defined(__clang_major__) && __clang_major__ >= 23
  i32x8 z8 = {0, 0, 0, 0, 0, 0, 0, 0};
  __builtin_amdgcn_tensor_load_to_lds(g0, g1, z4, z4, z8, 0);
#else
  __builtin_amdgcn_tensor_load_to_lds(g0, g1, z4, z4, 0);
#endif
}
#endif  // USE_TDM

// ---------------------------------------------------------------------------
// Kernel 1: build conv kernel c = cxx + cid, split bf16 hi/lo, tap-major K.
// ---------------------------------------------------------------------------
__global__ void build_c_split(const float* __restrict__ wxx,
                              const float* __restrict__ wid,
                              const float* __restrict__ XX,
                              const float* __restrict__ ident,
                              const int*   __restrict__ indsxx,
                              const int*   __restrict__ indsid,
                              int Lm1, int L,
                              bf16_t* __restrict__ a_hi,
                              bf16_t* __restrict__ a_lo) {
  int idx = blockIdx.x * blockDim.x + threadIdx.x;
  const int total = OUT_CH2 * IN_CH2 * KTAPS;
  if (idx >= total) return;
  int p  = idx % KTAPS;
  int i2 = (idx / KTAPS) % IN_CH2;
  int o2 = idx / (KTAPS * IN_CH2);
  int o = o2 >> 1, i = i2 >> 1;
  float cxx = wxx[(o * 64 + i) * Lm1 + indsxx[p]] * XX[(o2 * IN_CH2 + i2) * KTAPS + p];
  float cid = wid[(o * 64 + i) * L   + indsid[p]] * ident[o2 * IN_CH2 + i2];
  float c = cxx + cid;
  bf16_t hi = f2bf(c);
  bf16_t lo = f2bf(c - (float)hi);
  int k = p * IN_CH2 + i2;                      // tap-major K
  a_hi[(size_t)o2 * KDIM + k] = hi;
  a_lo[(size_t)o2 * KDIM + k] = lo;
}

// ---------------------------------------------------------------------------
// Kernel 2: implicit-GEMM conv, double-buffered, TDM-staged A.
// ---------------------------------------------------------------------------
__global__ void __launch_bounds__(256)
conv_wmma_bf16x2(const float* __restrict__ x,
                 const bf16_t* __restrict__ a_hi,
                 const bf16_t* __restrict__ a_lo,
                 float* __restrict__ out) {
  __shared__ SMem sm;

  const int blk = blockIdx.x;
  const int wb  = blk & 1;
  const int h   = (blk >> 1) & (HH - 1);
  const int b   = blk >> 8;
  const int w0  = wb * TILE_N;

  const int tid  = threadIdx.x;
  const int lane = tid & 31;
  const int wave = tid >> 5;
  const int wave_m = wave & 3;   // 32-row Co block
  const int wave_n = wave >> 2;  // 32-col pixel block

  // B-stage assignment: thread -> (pixel bn, ci-octet bk0)
  const int bn  = tid >> 2;
  const int bk0 = (tid & 3) * 8;

  // Fragment addressing (ISA 16-bit layouts)
  const int arow  = lane & 15;
  const int akoff = (lane < 16) ? 0 : 8;
  const int bkoff = (lane < 16) ? 0 : 8;
  const int bcol  = lane & 15;

  const v8f vzero = {0.f, 0.f, 0.f, 0.f, 0.f, 0.f, 0.f, 0.f};
  v8f acc[2][2];
  acc[0][0] = vzero; acc[0][1] = vzero; acc[1][0] = vzero; acc[1][1] = vzero;

  // per-chunk x address for this thread (tap-major: one tap per chunk)
  auto x_addr = [&](int kcs, bool& ok) -> const float* {
    int t   = kcs >> 2;
    int cib = (kcs & 3) << 5;
    int kh  = t / 5;
    int kw  = t - kh * 5;
    int hy  = h + kh - 2;
    int wx  = w0 + bn + kw - 2;
    ok = ((unsigned)hy < (unsigned)HH) & ((unsigned)wx < (unsigned)WW);
    return x + (((size_t)b * IN_CH2 + (cib + bk0)) * HH + hy) * WW + wx;
  };

  // stage one B chunk (64 px x 32 ci) into LDS buffer `buf`, branchless OOB
  auto stage_B = [&](int kcs, int buf) {
    bool ok;
    const float* p  = x_addr(kcs, ok);
    const float* ps = ok ? p : x;                // clamped safe base (2 cndmask)
    float v[8];
#pragma unroll
    for (int j = 0; j < 8; ++j) v[j] = ps[(size_t)j * HW];  // clause of 8 loads
#pragma unroll
    for (int j = 0; j < 8; ++j) v[j] = ok ? v[j] : 0.0f;    // cndmask to zero
    union { bf16_t e[8]; uint4 q; } uh, ul;
#pragma unroll
    for (int j = 0; j < 8; ++j) {
      bf16_t hv = f2bf(v[j]);
      uh.e[j] = hv;
      ul.e[j] = f2bf(v[j] - (float)hv);
    }
    *(uint4*)&sm.Bh[buf][bn * BSTRIDE + bk0] = uh.q;
    *(uint4*)&sm.Bl[buf][bn * BSTRIDE + bk0] = ul.q;
    // one-chunk-lookahead prefetch -> global_prefetch_b8
    if (kcs + 1 < NCHUNK) {
      bool ok2;
      const float* p2 = x_addr(kcs + 1, ok2);
      if (ok2) __builtin_prefetch(p2, 0, 0);
    }
  };

#if USE_TDM
  auto stage_A = [&](int kcs, int buf) {
    const size_t goff = (size_t)kcs * KC;
    tdm_load_a(a_hi + goff,
               (u32)(offsetof(SMem, Ah) + (size_t)buf * OUT_CH2 * BSTRIDE * sizeof(bf16_t)));
    tdm_load_a(a_lo + goff,
               (u32)(offsetof(SMem, Al) + (size_t)buf * OUT_CH2 * BSTRIDE * sizeof(bf16_t)));
  };
#endif

  // prologue: stage chunk 0
  stage_B(0, 0);
#if USE_TDM
  if (wave == 0) stage_A(0, 0);
#endif

  for (int kc = 0; kc < NCHUNK; ++kc) {
    const int cur = kc & 1;
    const int nxt = cur ^ 1;
#if USE_TDM
    if (wave == 0) __builtin_amdgcn_s_wait_tensorcnt(0);  // A(kc) in LDS
#endif
    __syncthreads();                                      // B(kc) visible too

    if (kc + 1 < NCHUNK) {
#if USE_TDM
      if (wave == 0) stage_A(kc + 1, nxt);
#endif
      stage_B(kc + 1, nxt);
    }

    // consume chunk kc: 12 WMMAs per wave
#pragma unroll
    for (int m2 = 0; m2 < 2; ++m2) {
      const int co = wave_m * 32 + m2 * 16 + arow;
#if USE_TDM
      const bf16_t* ph = &sm.Ah[cur][co * BSTRIDE + akoff];
      const bf16_t* pl = &sm.Al[cur][co * BSTRIDE + akoff];
#else
      const size_t agk = (size_t)kc * KC + akoff;
      const bf16_t* ph = a_hi + (size_t)co * KDIM + agk;
      const bf16_t* pl = a_lo + (size_t)co * KDIM + agk;
#endif
      v16bf Ah = load8x2(ph, ph + 16);
      v16bf Al = load8x2(pl, pl + 16);
#pragma unroll
      for (int n2 = 0; n2 < 2; ++n2) {
        const int col = wave_n * 32 + n2 * 16 + bcol;
        const bf16_t* qh = &sm.Bh[cur][col * BSTRIDE + bkoff];
        const bf16_t* ql = &sm.Bl[cur][col * BSTRIDE + bkoff];
        v16bf Bhf = load8x2(qh, qh + 16);
        v16bf Blf = load8x2(ql, ql + 16);
        acc[m2][n2] = __builtin_amdgcn_wmma_f32_16x16x32_bf16(
            false, Ah, false, Bhf, (short)0, acc[m2][n2], false, false);
        acc[m2][n2] = __builtin_amdgcn_wmma_f32_16x16x32_bf16(
            false, Ah, false, Blf, (short)0, acc[m2][n2], false, false);
        acc[m2][n2] = __builtin_amdgcn_wmma_f32_16x16x32_bf16(
            false, Al, false, Bhf, (short)0, acc[m2][n2], false, false);
      }
    }
  }

  // epilogue: D layout -> NCHW fp32 store
  const int dn   = lane & 15;
  const int mofs = (lane >> 4) * 8;
#pragma unroll
  for (int m2 = 0; m2 < 2; ++m2) {
#pragma unroll
    for (int n2 = 0; n2 < 2; ++n2) {
      int co_base = wave_m * 32 + m2 * 16 + mofs;
      int w = w0 + wave_n * 32 + n2 * 16 + dn;
#pragma unroll
      for (int r = 0; r < 8; ++r) {
        int co = co_base + r;
        out[(((size_t)b * OUT_CH2 + co) * HH + h) * WW + w] = acc[m2][n2][r];
      }
    }
  }
}

// ---------------------------------------------------------------------------
extern "C" void kernel_launch(void* const* d_in, const int* in_sizes, int n_in,
                              void* d_out, int out_size, void* d_ws, size_t ws_size,
                              hipStream_t stream) {
  const float* x      = (const float*)d_in[0];
  const float* wxx    = (const float*)d_in[1];
  const float* wid    = (const float*)d_in[2];
  const float* XX     = (const float*)d_in[3];
  const float* ident  = (const float*)d_in[4];
  const int*   indsxx = (const int*)d_in[5];
  const int*   indsid = (const int*)d_in[6];

  const int Lm1 = in_sizes[1] / (64 * 64);
  const int L   = in_sizes[2] / (64 * 64);

  bf16_t* a_hi = (bf16_t*)d_ws;
  bf16_t* a_lo = a_hi + (size_t)OUT_CH2 * KDIM;

  const int total_c = OUT_CH2 * IN_CH2 * KTAPS;
  build_c_split<<<(total_c + 255) / 256, 256, 0, stream>>>(
      wxx, wid, XX, ident, indsxx, indsid, Lm1, L, a_hi, a_lo);

  const int grid = BB * HH * (WW / TILE_N);
  conv_wmma_bf16x2<<<grid, 256, 0, stream>>>(x, a_hi, a_lo, (float*)d_out);
}
